// ExpCausalSelfAttention_36155034698422
// MI455X (gfx1250) — matrix operand order repaired and checked by reading, not compiled, not run on hardware
//
#include <hip/hip_runtime.h>

#define Bn 2
#define Tn 2048
#define Cn 1024
#define NH 16
#define NKV 4
#define HS 64

typedef __attribute__((ext_vector_type(16))) __bf16   v16bf;
typedef __attribute__((ext_vector_type(8)))  float    v8f;
typedef __attribute__((ext_vector_type(4)))  unsigned u32x4;
typedef __attribute__((ext_vector_type(8)))  int      i32x8;
typedef __attribute__((ext_vector_type(4)))  int      i32x4;

#define HAS_TDM __has_builtin(__builtin_amdgcn_tensor_load_to_lds)

// --- WMMA fragment index helpers (CDNA5 ISA 7.12.2 VGPR layouts) ---
__device__ __forceinline__ int a_kof(int i, int h) {
  int vg = i >> 1;
  return ((vg >> 2) << 4) + (h << 3) + ((vg & 3) << 1) + (i & 1);
}
__device__ __forceinline__ int b_kof(int i, int h) {
  int vg = i >> 1;
  return ((vg >> 2) << 4) + (h << 3) + ((vg & 3) << 1) + (i & 1);
}
// C/D 16x16 f32: vgpr r, lane l -> M = r + 8*(l/16), N = l%16.

template <int N> __device__ __forceinline__ void wait_tensorcnt() {
#if __has_builtin(__builtin_amdgcn_s_wait_tensorcnt)
  __builtin_amdgcn_s_wait_tensorcnt((short)N);
#endif
}
__device__ __forceinline__ void wait_dscnt0() {
  asm volatile("s_wait_dscnt 0x0" ::: "memory");
}

#if HAS_TDM
// Build a D# (group0 + group1) for a 2-D tile load per ISA 08_async_tensor §8
// and issue TENSOR_LOAD_TO_LDS. 2-D only: groups 2/3 zero (VADDR2/3 pattern).
__device__ __forceinline__ void tdm_load_2d(unsigned ldsByteOff, const void* gptr,
                                            unsigned rows, unsigned rowElems,
                                            unsigned rowStrideElems, unsigned dszLog2)
{
  unsigned long long ga = (unsigned long long)(uintptr_t)gptr;
  unsigned w0  = 1u;                                            // count=1
  unsigned w1  = ldsByteOff;                                    // lds_addr
  unsigned w2  = (unsigned)ga;                                  // global_addr lo
  unsigned w3  = ((unsigned)(ga >> 32) & 0x01ffffffu) | (2u << 30); // ga hi | type=2
  unsigned w4  = dszLog2 << 16;                                 // wg_mask=0 | data_size
  unsigned w5  = (rowElems & 0xffffu) << 16;                    // tensor_dim0 lo
  unsigned w6  = (rowElems >> 16) | ((rows & 0xffffu) << 16);   // td0 hi | tensor_dim1 lo
  unsigned w7  = (rows >> 16) | ((rowElems & 0xffffu) << 16);   // td1 hi | tile_dim0
  unsigned w8  = rows & 0xffffu;                                // tile_dim1 | tile_dim2=0
  unsigned w9  = rowStrideElems;                                // tensor_dim0_stride lo
  unsigned w10 = 0, w11 = 0;
#define RFL(x) (__builtin_amdgcn_readfirstlane((int)(x)))
  u32x4 g0 = { (unsigned)RFL(w0), (unsigned)RFL(w1), (unsigned)RFL(w2), (unsigned)RFL(w3) };
  i32x8 g1 = { RFL(w4), RFL(w5), RFL(w6), RFL(w7), RFL(w8), RFL(w9), RFL(w10), RFL(w11) };
#undef RFL
  i32x4 z4 = {0, 0, 0, 0};
#if __clang_major__ >= 23
  i32x8 z8 = {0, 0, 0, 0, 0, 0, 0, 0};
  __builtin_amdgcn_tensor_load_to_lds(g0, g1, z4, z4, z8, 0);
#else
  __builtin_amdgcn_tensor_load_to_lds(g0, g1, z4, z4, 0);
#endif
}
#endif

// =====================================================================
// Kernel 1: fused QKV projection, out = x @ W^T.
// Per wave: one 16x16 tile; operands TDM-staged (16x32 f32 tiles, double buf).
// Per-wave LDS: 2 bufs * (A 2KB + B 2KB) = 8KB; block = 64KB dynamic LDS.
// =====================================================================
#define GEMM_SMEM 65536

__global__ void qkv_wmma_kernel(const float* __restrict__ x,
                                const float* __restrict__ Wq,
                                const float* __restrict__ Wk,
                                const float* __restrict__ Wv,
                                float* __restrict__ qf,
                                float* __restrict__ kf,
                                float* __restrict__ vf)
{
  extern __shared__ char smem[];
  const int lane = threadIdx.x & 31;
  const int wave = threadIdx.x >> 5;
  const int row0 = blockIdx.x << 4;
  const int colTile = blockIdx.y * 8 + wave;
  const int h  = lane >> 4;
  const int mn = lane & 15;

  const float* W; int feat0; int which;
  if (colTile < 64)      { W = Wq; feat0 = colTile << 4;        which = 0; }
  else if (colTile < 80) { W = Wk; feat0 = (colTile - 64) << 4; which = 1; }
  else                   { W = Wv; feat0 = (colTile - 80) << 4; which = 2; }

  v8f acc = {};
#if HAS_TDM
  const unsigned ldsBase = (unsigned)__builtin_amdgcn_groupstaticsize()
                         + (unsigned)wave * 8192u;
  char* mybuf = smem + (size_t)wave * 8192;
  tdm_load_2d(ldsBase,         x + (size_t)row0  * Cn, 16, 32, Cn, 2);
  tdm_load_2d(ldsBase + 2048u, W + (size_t)feat0 * Cn, 16, 32, Cn, 2);
  for (int k0 = 0; k0 < Cn; k0 += 32) {
    const int buf = (k0 >> 5) & 1;
    if (k0 + 32 < Cn) {
      wait_dscnt0();
      tdm_load_2d(ldsBase + (unsigned)(buf ^ 1) * 4096u,
                  x + (size_t)row0 * Cn + k0 + 32, 16, 32, Cn, 2);
      tdm_load_2d(ldsBase + (unsigned)(buf ^ 1) * 4096u + 2048u,
                  W + (size_t)feat0 * Cn + k0 + 32, 16, 32, Cn, 2);
      wait_tensorcnt<2>();          // current pair landed, next pair in flight
    } else {
      wait_tensorcnt<0>();
    }
    const float* tA = (const float*)(mybuf + buf * 4096);
    const float* tB = tA + 512;
    v16bf a, bm;
#pragma unroll
    for (int i = 0; i < 16; ++i) {
      a[i]  = (__bf16)tA[mn * 32 + a_kof(i, h)];
      bm[i] = (__bf16)tB[mn * 32 + b_kof(i, h)];
    }
    acc = __builtin_amdgcn_wmma_f32_16x16x32_bf16(false, a, false, bm,
                                                  (short)0, acc, false, false);
  }
#else
  for (int k0 = 0; k0 < Cn; k0 += 32) {
    const float* xr = x + (size_t)(row0 + mn) * Cn + k0;
    const float* wr = W + (size_t)(feat0 + mn) * Cn + k0;
    v16bf a, bm;
#pragma unroll
    for (int i = 0; i < 16; ++i) {
      a[i]  = (__bf16)xr[a_kof(i, h)];
      bm[i] = (__bf16)wr[b_kof(i, h)];
    }
    acc = __builtin_amdgcn_wmma_f32_16x16x32_bf16(false, a, false, bm,
                                                  (short)0, acc, false, false);
  }
#endif

#pragma unroll
  for (int r = 0; r < 8; ++r) {
    int m = r + (h << 3);
    int grow = row0 + m;
    int bb = grow / Tn, tt = grow & (Tn - 1);
    int feat = feat0 + mn;
    int hd = feat >> 6, d = feat & 63;
    float v = acc[r];
    if (which == 0)      qf[(((size_t)bb * NH  + hd) * Tn + tt) * HS + d] = v;
    else if (which == 1) kf[(((size_t)bb * NKV + hd) * Tn + tt) * HS + d] = v;
    else                 vf[(((size_t)bb * NKV + hd) * Tn + tt) * HS + d] = v;
  }
}

// =====================================================================
// RoPE + bf16 convert (head-major keeps even/odd dim pairs adjacent)
// =====================================================================
__global__ void rope_bf16_kernel(const float* __restrict__ src,
                                 __bf16* __restrict__ dst, int totalPairs)
{
  int i = blockIdx.x * blockDim.x + threadIdx.x;
  if (i >= totalPairs) return;
  int d2 = i & 31;
  int t  = (i >> 5) & (Tn - 1);
  float freq = __expf(-0.28782313f * (float)d2);   // 10000^(-2*d2/64)
  float ang = (float)t * freq;
  float c = __cosf(ang), s = __sinf(ang);
  float e = src[2 * i], o = src[2 * i + 1];
  dst[2 * i]     = (__bf16)(e * c - o * s);
  dst[2 * i + 1] = (__bf16)(o * c + e * s);
}

__global__ void tobf16_kernel(const float* __restrict__ src,
                              __bf16* __restrict__ dst, int n)
{
  int i = blockIdx.x * blockDim.x + threadIdx.x;
  if (i < n) dst[i] = (__bf16)src[i];
}

// =====================================================================
// Kernel 3: attention. Block = (b, h, 16-query tile); 8 waves split keys.
// Dynamic LDS layout (bytes):
//   sS 0..131072 | sMax +1KB | sSum +1KB | sY +4KB | sQ +2KB | sKV 8*4KB
// =====================================================================
#define OFF_MAX  131072
#define OFF_SUM  132096
#define OFF_SY   133120
#define OFF_SQ   137216
#define OFF_SKV  139264
#define ATTN_SMEM 172032

__global__ void attn_wmma_kernel(const __bf16* __restrict__ qb,
                                 const __bf16* __restrict__ kb,
                                 const __bf16* __restrict__ vb,
                                 float* __restrict__ att,
                                 float* __restrict__ yh)
{
  extern __shared__ char smem[];
  float* sS   = (float*)smem;
  float* sMax = (float*)(smem + OFF_MAX);
  float* sSum = (float*)(smem + OFF_SUM);
  float* sY   = (float*)(smem + OFF_SY);

  const int tid  = threadIdx.x;
  const int lane = tid & 31;
  const int wave = tid >> 5;
  const int qtile = blockIdx.x & 127;
  const int bh    = blockIdx.x >> 7;
  const int bb  = bh >> 4;
  const int hh  = bh & 15;
  const int kvh = hh >> 2;
  const int q0  = qtile << 4;
  const int h2  = lane >> 4;
  const int mn  = lane & 15;

  const __bf16* Q = qb + (((size_t)bb * NH  + hh ) * Tn + q0) * HS;
  const __bf16* K = kb + ((size_t)bb * NKV + kvh) * Tn * HS;
  const __bf16* V = vb + ((size_t)bb * NKV + kvh) * Tn * HS;

#if HAS_TDM
  const unsigned ldsBase = (unsigned)__builtin_amdgcn_groupstaticsize();
  const unsigned myKVoff = ldsBase + OFF_SKV + (unsigned)wave * 4096u;
  char* myKV = smem + OFF_SKV + (size_t)wave * 4096;
  if (wave == 0) tdm_load_2d(ldsBase + OFF_SQ, Q, 16, 64, 64, 1);
#endif

  for (int i = tid; i < 16 * Tn; i += 256) sS[i] = -1e30f;   // masked default
#if HAS_TDM
  if (wave == 0) wait_tensorcnt<0>();
#endif
  __syncthreads();

  // hoist the shared Q fragments (A-matrix, 16x64 = two k-chunks)
  v16bf aQ[2];
#if HAS_TDM
  {
    const __bf16* sQ = (const __bf16*)(smem + OFF_SQ);
#pragma unroll
    for (int kc2 = 0; kc2 < 2; ++kc2)
#pragma unroll
      for (int i = 0; i < 16; ++i)
        aQ[kc2][i] = sQ[mn * HS + kc2 * 32 + a_kof(i, h2)];
  }
#else
#pragma unroll
  for (int kc2 = 0; kc2 < 2; ++kc2)
#pragma unroll
    for (int i = 0; i < 16; ++i)
      aQ[kc2][i] = Q[(size_t)mn * HS + kc2 * 32 + a_kof(i, h2)];
#endif

  // ---- Phase 1: S = (1/8) Q K^T, key tiles on/below diagonal ----
#if HAS_TDM
  if (wave <= qtile)
    tdm_load_2d(myKVoff, K + (size_t)wave * 16 * HS, 16, 64, 64, 1);
  int idx = 0;
  for (int kt = wave; kt <= qtile; kt += 8, ++idx) {
    const int buf = idx & 1;
    if (kt + 8 <= qtile) {
      wait_dscnt0();
      tdm_load_2d(myKVoff + (unsigned)(buf ^ 1) * 2048u,
                  K + (size_t)(kt + 8) * 16 * HS, 16, 64, 64, 1);
      wait_tensorcnt<1>();
    } else {
      wait_tensorcnt<0>();
    }
    const __bf16* tK = (const __bf16*)(myKV + buf * 2048);
    v8f acc = {};
#pragma unroll
    for (int kc2 = 0; kc2 < 2; ++kc2) {
      v16bf bm;
#pragma unroll
      for (int i = 0; i < 16; ++i)
        bm[i] = tK[mn * HS + kc2 * 32 + b_kof(i, h2)];
      acc = __builtin_amdgcn_wmma_f32_16x16x32_bf16(false, aQ[kc2], false, bm,
                                                    (short)0, acc, false, false);
    }
    const int key0 = kt << 4;
#pragma unroll
    for (int r = 0; r < 8; ++r) {
      int m = r + (h2 << 3);
      int col = key0 + mn;
      float v = acc[r] * 0.125f;
      if (col > q0 + m) v = -1e30f;
      sS[m * Tn + col] = v;
    }
  }
#else
  for (int kt = wave; kt <= qtile; kt += 8) {
    const int key0 = kt << 4;
    v8f acc = {};
#pragma unroll
    for (int kc2 = 0; kc2 < 2; ++kc2) {
      v16bf bm;
#pragma unroll
      for (int i = 0; i < 16; ++i)
        bm[i] = K[(size_t)(key0 + mn) * HS + kc2 * 32 + b_kof(i, h2)];
      acc = __builtin_amdgcn_wmma_f32_16x16x32_bf16(false, aQ[kc2], false, bm,
                                                    (short)0, acc, false, false);
    }
#pragma unroll
    for (int r = 0; r < 8; ++r) {
      int m = r + (h2 << 3);
      int col = key0 + mn;
      float v = acc[r] * 0.125f;
      if (col > q0 + m) v = -1e30f;
      sS[m * Tn + col] = v;
    }
  }
#endif
  __syncthreads();

  // ---- Phase 2: softmax over 2048 keys; write att; probs stay in LDS ----
  const int r    = tid >> 4;
  const int cseg = tid & 15;
  const int c0   = cseg * (Tn / 16);
  float lm = -1e30f;
  for (int c = 0; c < Tn / 16; ++c) lm = fmaxf(lm, sS[r * Tn + c0 + c]);
  sMax[r * 16 + cseg] = lm;
  __syncthreads();
  float rm = -1e30f;
#pragma unroll
  for (int j = 0; j < 16; ++j) rm = fmaxf(rm, sMax[r * 16 + j]);
  float ls = 0.f;
  for (int c = 0; c < Tn / 16; ++c) {
    float e = __expf(sS[r * Tn + c0 + c] - rm);
    sS[r * Tn + c0 + c] = e;
    ls += e;
  }
  sSum[r * 16 + cseg] = ls;
  __syncthreads();
  float rs = 0.f;
#pragma unroll
  for (int j = 0; j < 16; ++j) rs += sSum[r * 16 + j];
  float inv = 1.0f / rs;
  float* attRow = att + ((size_t)bh * Tn + q0 + r) * Tn;
  for (int c = 0; c < Tn / 16; ++c) {
    float pv = sS[r * Tn + c0 + c] * inv;
    sS[r * Tn + c0 + c] = pv;
    attRow[c0 + c] = pv;
  }
  for (int i = tid; i < 16 * HS; i += 256) sY[i] = 0.f;
  __syncthreads();

  // ---- Phase 3: y = P @ V over the wave's key slice ----
  v8f accd[4] = {};
  const int k0f  = wave * (Tn / 8);
  const int kEnd = (k0f + Tn / 8 < q0 + 16) ? (k0f + Tn / 8) : (q0 + 16);
#if HAS_TDM
  if (k0f < kEnd) {
    wait_dscnt0();
    tdm_load_2d(myKVoff, V + (size_t)k0f * HS, 32, 64, 64, 1);
  }
  for (int k0 = k0f; k0 < kEnd; k0 += 32) {
    wait_tensorcnt<0>();
    const __bf16* tV = (const __bf16*)myKV;
    v16bf a;
#pragma unroll
    for (int i = 0; i < 16; ++i)
      a[i] = (__bf16)sS[mn * Tn + k0 + a_kof(i, h2)];
    v16bf bmv[4];
#pragma unroll
    for (int dt = 0; dt < 4; ++dt)
#pragma unroll
      for (int i = 0; i < 16; ++i)
        bmv[dt][i] = tV[b_kof(i, h2) * HS + (dt << 4) + mn];
    wait_dscnt0();
    if (k0 + 32 < kEnd)
      tdm_load_2d(myKVoff, V + (size_t)(k0 + 32) * HS, 32, 64, 64, 1);
#pragma unroll
    for (int dt = 0; dt < 4; ++dt)
      accd[dt] = __builtin_amdgcn_wmma_f32_16x16x32_bf16(false, a, false, bmv[dt],
                                                         (short)0, accd[dt],
                                                         false, false);
  }
#else
  for (int k0 = k0f; k0 < kEnd; k0 += 32) {
    v16bf a;
#pragma unroll
    for (int i = 0; i < 16; ++i)
      a[i] = (__bf16)sS[mn * Tn + k0 + a_kof(i, h2)];
#pragma unroll
    for (int dt = 0; dt < 4; ++dt) {
      v16bf bm;
#pragma unroll
      for (int i = 0; i < 16; ++i)
        bm[i] = V[(size_t)(k0 + b_kof(i, h2)) * HS + (dt << 4) + mn];
      accd[dt] = __builtin_amdgcn_wmma_f32_16x16x32_bf16(false, a, false, bm,
                                                         (short)0, accd[dt],
                                                         false, false);
    }
  }
#endif
#pragma unroll
  for (int dt = 0; dt < 4; ++dt)
#pragma unroll
    for (int rr = 0; rr < 8; ++rr) {
      int m = rr + (h2 << 3);
      atomicAdd(&sY[m * HS + (dt << 4) + mn], accd[dt][rr]);  // ds_add_f32
    }
  __syncthreads();

  for (int i = tid; i < 16 * HS; i += 256) {
    int m = i >> 6, d = i & 63;
    yh[((size_t)bb * Tn + q0 + m) * Cn + hh * HS + d] = sY[i];
  }
}

// =====================================================================
// Kernel 4: output projection y = yh @ Wproj^T + b (TDM-staged like K1)
// =====================================================================
__global__ void proj_wmma_kernel(const float* __restrict__ yh,
                                 const float* __restrict__ Wp,
                                 const float* __restrict__ bias,
                                 float* __restrict__ out)
{
  extern __shared__ char smem[];
  const int lane = threadIdx.x & 31;
  const int wave = threadIdx.x >> 5;
  const int row0 = blockIdx.x << 4;
  const int col0 = (blockIdx.y * 8 + wave) << 4;
  const int h  = lane >> 4;
  const int mn = lane & 15;

  v8f acc = {};
#if HAS_TDM
  const unsigned ldsBase = (unsigned)__builtin_amdgcn_groupstaticsize()
                         + (unsigned)wave * 8192u;
  char* mybuf = smem + (size_t)wave * 8192;
  tdm_load_2d(ldsBase,         yh + (size_t)row0 * Cn, 16, 32, Cn, 2);
  tdm_load_2d(ldsBase + 2048u, Wp + (size_t)col0 * Cn, 16, 32, Cn, 2);
  for (int k0 = 0; k0 < Cn; k0 += 32) {
    const int buf = (k0 >> 5) & 1;
    if (k0 + 32 < Cn) {
      wait_dscnt0();
      tdm_load_2d(ldsBase + (unsigned)(buf ^ 1) * 4096u,
                  yh + (size_t)row0 * Cn + k0 + 32, 16, 32, Cn, 2);
      tdm_load_2d(ldsBase + (unsigned)(buf ^ 1) * 4096u + 2048u,
                  Wp + (size_t)col0 * Cn + k0 + 32, 16, 32, Cn, 2);
      wait_tensorcnt<2>();
    } else {
      wait_tensorcnt<0>();
    }
    const float* tA = (const float*)(mybuf + buf * 4096);
    const float* tB = tA + 512;
    v16bf a, bm;
#pragma unroll
    for (int i = 0; i < 16; ++i) {
      a[i]  = (__bf16)tA[mn * 32 + a_kof(i, h)];
      bm[i] = (__bf16)tB[mn * 32 + b_kof(i, h)];
    }
    acc = __builtin_amdgcn_wmma_f32_16x16x32_bf16(false, a, false, bm,
                                                  (short)0, acc, false, false);
  }
#else
  for (int k0 = 0; k0 < Cn; k0 += 32) {
    const float* ar = yh + (size_t)(row0 + mn) * Cn + k0;
    const float* wr = Wp + (size_t)(col0 + mn) * Cn + k0;
    v16bf a, bm;
#pragma unroll
    for (int i = 0; i < 16; ++i) {
      a[i]  = (__bf16)ar[a_kof(i, h)];
      bm[i] = (__bf16)wr[b_kof(i, h)];
    }
    acc = __builtin_amdgcn_wmma_f32_16x16x32_bf16(false, a, false, bm,
                                                  (short)0, acc, false, false);
  }
#endif
#pragma unroll
  for (int r = 0; r < 8; ++r) {
    int m = r + (h << 3);
    out[(size_t)(row0 + m) * Cn + col0 + mn] = acc[r] + bias[col0 + mn];
  }
}

// =====================================================================
extern "C" void kernel_launch(void* const* d_in, const int* in_sizes, int n_in,
                              void* d_out, int out_size, void* d_ws, size_t ws_size,
                              hipStream_t stream)
{
  (void)in_sizes; (void)n_in; (void)out_size; (void)ws_size;
  const float* x     = (const float*)d_in[0];
  const float* Wq    = (const float*)d_in[1];
  const float* Wk    = (const float*)d_in[2];
  const float* Wv    = (const float*)d_in[3];
  const float* Wproj = (const float*)d_in[4];
  const float* bproj = (const float*)d_in[5];

  float* y_out   = (float*)d_out;
  float* att_out = y_out + (size_t)Bn * Tn * Cn;

  const size_t Qe = (size_t)Bn * NH  * Tn * HS;
  const size_t Ke = (size_t)Bn * NKV * Tn * HS;

  char* p = (char*)d_ws;
  float*  qf  = (float*)p;  p += Qe * 4;
  float*  kf  = (float*)p;  p += Ke * 4;
  float*  vf  = (float*)p;  p += Ke * 4;
  __bf16* qbf = (__bf16*)p; p += Qe * 2;
  __bf16* kbf = (__bf16*)p; p += Ke * 2;
  __bf16* vbf = (__bf16*)p; p += Ke * 2;
  float*  yhf = (float*)p;  p += (size_t)Bn * Tn * Cn * 4;

  hipFuncSetAttribute((const void*)qkv_wmma_kernel,
                      hipFuncAttributeMaxDynamicSharedMemorySize, GEMM_SMEM);
  hipFuncSetAttribute((const void*)proj_wmma_kernel,
                      hipFuncAttributeMaxDynamicSharedMemorySize, GEMM_SMEM);
  hipFuncSetAttribute((const void*)attn_wmma_kernel,
                      hipFuncAttributeMaxDynamicSharedMemorySize, ATTN_SMEM);

  qkv_wmma_kernel<<<dim3((Bn * Tn) / 16, 12), 256, GEMM_SMEM, stream>>>(
      x, Wq, Wk, Wv, qf, kf, vf);

  rope_bf16_kernel<<<(int)((Qe / 2 + 255) / 256), 256, 0, stream>>>(qf, qbf, (int)(Qe / 2));
  rope_bf16_kernel<<<(int)((Ke / 2 + 255) / 256), 256, 0, stream>>>(kf, kbf, (int)(Ke / 2));
  tobf16_kernel<<<(int)((Ke + 255) / 256), 256, 0, stream>>>(vf, vbf, (int)Ke);

  attn_wmma_kernel<<<Bn * NH * (Tn / 16), 256, ATTN_SMEM, stream>>>(
      qbf, kbf, vbf, att_out, yhf);

  proj_wmma_kernel<<<dim3((Bn * Tn) / 16, 8), 256, GEMM_SMEM, stream>>>(
      yhf, Wproj, bproj, y_out);
}
